// TinyRelu2NVFP4Linear_84670985273479
// MI455X (gfx1250) — matrix operand order repaired and checked by reading, not compiled, not run on hardware
//
#include <hip/hip_runtime.h>
#include <hip/hip_bf16.h>

typedef __attribute__((ext_vector_type(16))) _Float16 v16h;
typedef __attribute__((ext_vector_type(8)))  _Float16 v8h;
typedef __attribute__((ext_vector_type(8)))  float    v8f;

#define GEMM_M 128
#define GEMM_K 4096
#define GEMM_N 11008

// ---------- NVFP4 helpers ----------

// Round-to-nearest-even onto the fp8 e4m3fn grid (input >= 0), saturate at 448.
// step is a power of two -> reciprocal built exactly from bits, no division.
__device__ __forceinline__ float cast_e4m3(float x) {
    if (!(x > 0.f)) return 0.f;
    x = fminf(x, 448.f);
    unsigned bits = __float_as_uint(x);
    int e = (int)(bits >> 23) - 127;         // floor(log2 x) for normals
    int qe = e - 3;                          // 3 mantissa bits
    if (qe < -9) qe = -9;                    // e4m3 denormal quantum 2^-9
    float step  = __uint_as_float((unsigned)(qe + 127) << 23);  // 2^qe
    float rstep = __uint_as_float((unsigned)(127 - qe) << 23);  // 2^-qe (exact)
    return rintf(x * rstep) * step;          // RNE on the grid
}

// fp4 e2m1 grid {0,.5,1,1.5,2,3,4,6}; ties toward larger magnitude, matching
// jnp.searchsorted side='right'. Grid step by binade: 0.5 / 1 / 2.
// floor(a*rs + 0.5): all midpoints are exact in f32, so tie direction is exact.
__device__ __forceinline__ float cast_e2m1(float x) {
    float a = fminf(fabsf(x), 6.f);
    float rs = (a < 2.f) ? 2.0f : ((a < 4.f) ? 1.f : 0.5f);
    float s  = (a < 2.f) ? 0.5f : ((a < 4.f) ? 1.f : 2.0f);
    float q = floorf(__builtin_fmaf(a, rs, 0.5f)) * s;
    return copysignf(q, x);
}

// ---------- Pass 0: init workspace scalar ----------
__global__ void k_init(unsigned* amax_bits) { *amax_bits = 0u; }

// ---------- Pass 1: global amax of |weight| (bit-compare atomicMax) ----------
__global__ void __launch_bounds__(256)
k_amax(const float* __restrict__ w, unsigned* __restrict__ amax_bits) {
    const size_t n4 = (size_t)GEMM_N * GEMM_K / 4;
    size_t stride = (size_t)gridDim.x * blockDim.x;
    float m = 0.f;
    for (size_t i = (size_t)blockIdx.x * blockDim.x + threadIdx.x; i < n4; i += stride) {
        float4 v = ((const float4*)w)[i];
        m = fmaxf(m, fmaxf(fmaxf(fabsf(v.x), fabsf(v.y)),
                           fmaxf(fabsf(v.z), fabsf(v.w))));
    }
    __shared__ float red[256];
    red[threadIdx.x] = m;
    __syncthreads();
    for (int s = 128; s > 0; s >>= 1) {
        if (threadIdx.x < (unsigned)s)
            red[threadIdx.x] = fmaxf(red[threadIdx.x], red[threadIdx.x + s]);
        __syncthreads();
    }
    if (threadIdx.x == 0) atomicMax(amax_bits, __float_as_uint(red[0]));
}

// ---------- Pass 2: ReLU^2 + NVFP4 quantize x, write f16 dequant values ----------
// Output layout = WMMA A-fragment order: xw[mt][ksg][lane][e] halves, where
// 16-bit A 16x32: lane = (m%16) + 16*half, half=(kk>>3)&1, e=(kk&7)+8*(kk>>4).
__global__ void __launch_bounds__(256)
k_quant_x(const float* __restrict__ x, const float* __restrict__ gscale,
          _Float16* __restrict__ xw) {
    const int K = GEMM_K, KSTEPS = GEMM_K / 32;
    int idx = blockIdx.x * blockDim.x + threadIdx.x;  // one thread per 16-block
    int m  = idx >> 8;          // K/16 = 256 blocks per row
    int kb = idx & 255;
    float gs = gscale[0];

    const float* p = x + (size_t)m * K + (size_t)kb * 16;
    float v[16], amax = 0.f;
#pragma unroll
    for (int i = 0; i < 16; ++i) {
        float t = fmaxf(p[i], 0.f);
        t = t * t;                      // ReLU^2
        v[i] = t;
        amax = fmaxf(amax, t);
    }
    float sf = cast_e4m3(amax * (1.f / 6.f) * gs);
    float inv = (sf > 0.f) ? gs / sf : 1.f;   // one divide per block (== 1/dq_scale)

    v8h lo, hi;                         // j=0..7 -> half0, j=8..15 -> half1
#pragma unroll
    for (int i = 0; i < 8; ++i) lo[i] = (_Float16)(cast_e2m1(v[i] * inv) * sf);
#pragma unroll
    for (int i = 0; i < 8; ++i) hi[i] = (_Float16)(cast_e2m1(v[8 + i] * inv) * sf);

    int mt = m >> 4, ks = kb >> 1, mrow = m & 15, odd = kb & 1;  // e-offset = odd*8
    size_t base = (size_t)(mt * KSTEPS + ks) * 32 * 16;
    *(v8h*)(xw + base + (size_t)mrow * 16 + odd * 8)        = lo;
    *(v8h*)(xw + base + (size_t)(mrow + 16) * 16 + odd * 8) = hi;
}

// ---------- Pass 3: fused weight NVFP4 quantize + WMMA GEMM + epilogue ----------
// Workgroup: 128 threads (4 waves), N-strip of 16, loops K in 128-wide chunks.
// Each thread quantizes one 16-block of weight per chunk into LDS in B-fragment
// order (lanes 0-15: K=0-7,16-23; lanes 16-31: K=8-15,24-31). Each wave owns 2 M-tiles.
__global__ void __launch_bounds__(128)
k_gemm(const float* __restrict__ w, const float* __restrict__ bias,
       const _Float16* __restrict__ xw, const float* __restrict__ gscale,
       const unsigned* __restrict__ amax_bits, float* __restrict__ out) {
    const int K = GEMM_K, N = GEMM_N, KSTEPS = GEMM_K / 32;
    __shared__ __align__(32) _Float16 lb[4 * 32 * 16];   // 4 k-steps x B-fragment

    float w_amax = __uint_as_float(*amax_bits);
    float w_gs  = (448.f * 6.f) / w_amax;                // fp4_global_scale(weight)
    float alpha = 1.f / (gscale[0] * w_gs);

    const int n0   = blockIdx.x * 16;
    const int tid  = threadIdx.x;
    const int wave = tid >> 5, lane = tid & 31;
    const int mt0 = wave * 2, mt1 = wave * 2 + 1;

    v8f acc0 = {}, acc1 = {};

    const int nl = tid >> 3;            // 0..15 : local row of weight
    const int kb = tid & 7;             // 0..7  : 16-block within 128-chunk
    const int ksl = kb >> 1, bhalf = kb & 1;
    const int lbase = ((ksl * 32) + nl) * 16 + 8 * bhalf;

    for (int kc = 0; kc < K / 128; ++kc) {
        // --- in-register NVFP4 quantize of one weight block ---
        const float* p = w + (size_t)(n0 + nl) * K + (size_t)kc * 128 + (size_t)kb * 16;
        float vv[16], amax = 0.f;
#pragma unroll
        for (int i = 0; i < 16; ++i) {
            float t = p[i];
            vv[i] = t;
            amax = fmaxf(amax, fabsf(t));
        }
        float sf = cast_e4m3(amax * (1.f / 6.f) * w_gs);
        float inv = (sf > 0.f) ? w_gs / sf : 1.f;  // one divide per block
        v8h h0, h1;
#pragma unroll
        for (int i = 0; i < 8; ++i) h0[i] = (_Float16)(cast_e2m1(vv[i] * inv) * sf);
#pragma unroll
        for (int i = 0; i < 8; ++i) h1[i] = (_Float16)(cast_e2m1(vv[8 + i] * inv) * sf);
        *(v8h*)(lb + lbase)     = h0;   // K = bhalf*16 + 0..7
        *(v8h*)(lb + lbase + 256) = h1;   // K = bhalf*16 + 8..15
        __syncthreads();

        // --- 4 WMMA k-steps over the staged chunk ---
#pragma unroll
        for (int ks = 0; ks < 4; ++ks) {
            v16h bfrag = *(const v16h*)(lb + (ks * 32 + lane) * 16);
            int ksg = kc * 4 + ks;
            v16h a0 = *(const v16h*)(xw + ((size_t)(mt0 * KSTEPS + ksg) * 32 + lane) * 16);
            v16h a1 = *(const v16h*)(xw + ((size_t)(mt1 * KSTEPS + ksg) * 32 + lane) * 16);
            acc0 = __builtin_amdgcn_wmma_f32_16x16x32_f16(
                       false, a0, false, bfrag, (short)0, acc0, false, false);
            acc1 = __builtin_amdgcn_wmma_f32_16x16x32_f16(
                       false, a1, false, bfrag, (short)0, acc1, false, false);
        }
        __syncthreads();
    }

    // --- epilogue: D layout 16x16 f32 -> lane = N + 16*half, vgpr r = M (+8*half)
    const int chalf = lane >> 4, nn = lane & 15;
    float bv = bias[n0 + nn];
#pragma unroll
    for (int r = 0; r < 8; ++r) {
        int m0 = mt0 * 16 + r + 8 * chalf;
        int m1 = mt1 * 16 + r + 8 * chalf;
        out[(size_t)m0 * N + n0 + nn] = acc0[r] * alpha + bv;
        out[(size_t)m1 * N + n0 + nn] = acc1[r] * alpha + bv;
    }
}

// ---------- launcher ----------
extern "C" void kernel_launch(void* const* d_in, const int* in_sizes, int n_in,
                              void* d_out, int out_size, void* d_ws, size_t ws_size,
                              hipStream_t stream) {
    const float* x      = (const float*)d_in[0];   // [128, 4096]
    const float* weight = (const float*)d_in[1];   // [11008, 4096]
    const float* bias   = (const float*)d_in[2];   // [11008]
    const float* iscale = (const float*)d_in[3];   // [1]
    float* out = (float*)d_out;                    // [128, 11008] fp32

    unsigned* amax_bits = (unsigned*)d_ws;
    _Float16* xw = (_Float16*)((char*)d_ws + 256); // 128*4096 halves = 1 MB

    k_init   <<<1, 1, 0, stream>>>(amax_bits);
    k_amax   <<<1024, 256, 0, stream>>>(weight, amax_bits);
    k_quant_x<<<(GEMM_M * (GEMM_K / 16)) / 256, 256, 0, stream>>>(x, iscale, xw);
    k_gemm   <<<GEMM_N / 16, 128, 0, stream>>>(weight, bias, xw, iscale, amax_bits, out);
}
